// DMP_50912542327270
// MI455X (gfx1250) — matrix operand order repaired and checked by reading, not compiled, not run on hardware
//
#include <hip/hip_runtime.h>
#include <hip/hip_bf16.h>
#include <math.h>
#include <stdint.h>

// ---------------------------------------------------------------------------
// MI455X (gfx1250) implementation.
// Dominant cost: 16 slots x 2 GEMMs of [16384x512]x[512x512] = ~275 GFLOP.
// All operands fit in 192MB L2 -> compute bound -> v_wmma_f32_16x16x32_bf16
// with fp32 accumulation. Tiles are staged into LDS by the Tensor Data Mover
// (tensor_load_to_lds, TENSORcnt), double-buffered so DMA overlaps WMMA.
// Logits-only first pass + top-k row recompute avoids the 512MB x tensor.
// ---------------------------------------------------------------------------

#define D     512
#define LSEQ  4096
#define NB    4
#define NMEM  16
#define ROWS  (NB * LSEQ)   // 16384
#define TOPK  8

typedef __bf16 bf16_t;
typedef __attribute__((ext_vector_type(16))) __bf16 v16bf;
typedef __attribute__((ext_vector_type(8)))  __bf16 v8bf;
typedef __attribute__((ext_vector_type(4)))  __bf16 v4bf;
typedef __attribute__((ext_vector_type(8)))  float  v8f;
typedef __attribute__((ext_vector_type(4)))  float  v4f;
typedef __attribute__((ext_vector_type(4)))  unsigned int u32x4;
typedef __attribute__((ext_vector_type(4)))  int   i32x4;
typedef __attribute__((ext_vector_type(8)))  int   i32x8;

#if defined(__has_builtin)
#if __has_builtin(__builtin_amdgcn_tensor_load_to_lds)
#define HAVE_TDM 1
#endif
#endif

__device__ __forceinline__ v16bf cat8(v8bf lo, v8bf hi) {
  v16bf r;
#pragma unroll
  for (int i = 0; i < 8; ++i) { r[i] = lo[i]; r[8 + i] = hi[i]; }
  return r;
}

#if HAVE_TDM
// 2D TDM load: `rows` rows of `row_units` 8-byte units, global row pitch
// `stride_units` (8B units), into contiguous LDS at lds_off.
__device__ __forceinline__ void tdm_load_2d(const void* gsrc, uint32_t lds_off,
                                            uint32_t rows, uint32_t row_units,
                                            uint32_t stride_units) {
  uint64_t ga = (uint64_t)(uintptr_t)gsrc;
  u32x4 g0;
  g0[0] = 1u;                                               // count=1 (valid), user mode
  g0[1] = lds_off;                                          // lds_addr (bytes)
  g0[2] = (uint32_t)(ga & 0xffffffffu);                     // global_addr[31:0]
  g0[3] = (uint32_t)((ga >> 32) & 0x1ffffffu) | (2u << 30); // addr[56:32] | type=2
  i32x8 g1;
  g1[0] = (int)(3u << 16);                                  // data_size=3 (8B), mask=0
  g1[1] = (int)((row_units & 0xffffu) << 16);               // tensor_dim0[15:0]
  g1[2] = (int)((row_units >> 16) | ((rows & 0xffffu) << 16)); // td0[31:16]|td1[15:0]
  g1[3] = (int)((rows >> 16) | ((row_units & 0xffffu) << 16)); // td1[31:16]|tile_dim0
  g1[4] = (int)(rows & 0xffffu);                            // tile_dim1 | tile_dim2=0
  g1[5] = (int)stride_units;                                // tensor_dim0_stride[31:0]
  g1[6] = 0;                                                // stride0[47:32]|stride1 lo
  g1[7] = 0;                                                // stride1 hi
  i32x4 z4 = {0, 0, 0, 0};
#if defined(__clang_major__) && __clang_major__ >= 23
  i32x8 z8 = {0, 0, 0, 0, 0, 0, 0, 0};
  __builtin_amdgcn_tensor_load_to_lds(g0, g1, z4, z4, z8, 0);
#else
  __builtin_amdgcn_tensor_load_to_lds(g0, g1, z4, z4, 0);
#endif
}
#endif

// ---------------- Pass 0a: query fp32 -> bf16 ------------------------------
__global__ __launch_bounds__(256)
void k_cvt_query(const float* __restrict__ q, bf16_t* __restrict__ qb) {
  size_t i = ((size_t)blockIdx.x * 256 + threadIdx.x) * 4;
  v4f v = *(const v4f*)(q + i);
  v4bf o;
#pragma unroll
  for (int t = 0; t < 4; ++t) o[t] = (__bf16)v[t];
  *(v4bf*)(qb + i) = o;
}

// ---------------- Pass 0b: transpose W1/W2, fp32 -> bf16 -------------------
// w1t[n][h][d] = W1[n][d][h]  (B-fragments need K contiguous per column)
__global__ __launch_bounds__(256)
void k_transpose_w(const float* __restrict__ W1, const float* __restrict__ W2,
                   bf16_t* __restrict__ w1t, bf16_t* __restrict__ w2t) {
  __shared__ float t[32][33];
  int mat  = blockIdx.z >> 4;
  int slot = blockIdx.z & 15;
  const float* W = (mat ? W2 : W1) + (size_t)slot * D * D;
  bf16_t*      O = (mat ? w2t : w1t) + (size_t)slot * D * D;
  int hb = blockIdx.x * 32, db = blockIdx.y * 32;
  int tx = threadIdx.x & 31, ty = threadIdx.x >> 5;   // ty: 0..7
#pragma unroll
  for (int rr = 0; rr < 4; ++rr) {
    int d = ty + rr * 8;
    t[d][tx] = W[(size_t)(db + d) * D + hb + tx];     // coalesced read
  }
  __syncthreads();
#pragma unroll
  for (int rr = 0; rr < 4; ++rr) {
    int h = ty + rr * 8;
    O[(size_t)(hb + h) * D + db + tx] = (__bf16)t[tx][h];  // coalesced write
  }
}

// ---------------- Pass A: fused GEMM1+GEMM2 -> logits ----------------------
// grid (256, 16): 64-row tile x slot. block 256 = 8 waves; wave w owns
// columns [w*64, w*64+64). 16 WMMA accumulators (4 Mfrag x 4 Nfrag) per wave.
__global__ __launch_bounds__(256)
void k_fused_logits(const bf16_t* __restrict__ qb,
                    const bf16_t* __restrict__ w1t,
                    const bf16_t* __restrict__ w2t,
                    const float*  __restrict__ b1,
                    const float*  __restrict__ b2,
                    const float*  __restrict__ Wg,
                    float* __restrict__ logits) {
  __shared__ alignas(64) bf16_t As[2][64 * 32];    //  2 x 4 KB A k-tile
  __shared__ alignas(64) bf16_t Bs[2][512 * 32];   //  2 x 32 KB B k-tile [N][32]
  __shared__ alignas(64) bf16_t Hs[64 * 512];      // 64 KB H tile [m][h]
  __shared__ float Lg[64];

  const int slot   = blockIdx.y;
  const int m_base = blockIdx.x * 64;
  const int tid    = threadIdx.x;
  const int wave   = tid >> 5;
  const int lane   = tid & 31;
  const int hb     = lane >> 4;       // half-wave select (ISA A/B layouts)
  const int l15    = lane & 15;
  const int n0     = wave * 64;
  const float slope = 0.01f + 0.0125f * (float)slot;

  const bf16_t* W1s = w1t + (size_t)slot * D * D;
  const bf16_t* W2s = w2t + (size_t)slot * D * D;

  v8f acc[4][4];
  auto zero_acc = [&]() {
#pragma unroll
    for (int a = 0; a < 4; ++a)
#pragma unroll
      for (int c = 0; c < 4; ++c)
#pragma unroll
        for (int r = 0; r < 8; ++r) acc[a][c][r] = 0.f;
  };

  // one 64x64x32 MACRO-step: A from Asb (64x32), B from Bsb ([N][32])
  auto mma_step = [&](const bf16_t* Asb, const bf16_t* Bsb) {
    v16bf af[4], bfm[4];
#pragma unroll
    for (int mf = 0; mf < 4; ++mf) {
      const bf16_t* p = Asb + (mf * 16 + l15) * 32 + hb * 16;
      af[mf] = cat8(*(const v8bf*)p, *(const v8bf*)(p + 8));
    }
#pragma unroll
    for (int jf = 0; jf < 4; ++jf)
      bfm[jf] = *(const v16bf*)(Bsb + (n0 + jf * 16 + l15) * 32 + hb * 16);
#pragma unroll
    for (int mf = 0; mf < 4; ++mf)
#pragma unroll
      for (int jf = 0; jf < 4; ++jf)
        acc[mf][jf] = __builtin_amdgcn_wmma_f32_16x16x32_bf16(
            false, af[mf], false, bfm[jf], (short)0, acc[mf][jf], false, false);
  };
  // same, but A comes straight from the (stable) Hs tile
  auto mma_step_h = [&](int kt, const bf16_t* Bsb) {
    v16bf af[4], bfm[4];
#pragma unroll
    for (int mf = 0; mf < 4; ++mf) {
      const bf16_t* p = Hs + (mf * 16 + l15) * D + kt * 32 + hb * 16;
      af[mf] = cat8(*(const v8bf*)p, *(const v8bf*)(p + 8));
    }
#pragma unroll
    for (int jf = 0; jf < 4; ++jf)
      bfm[jf] = *(const v16bf*)(Bsb + (n0 + jf * 16 + l15) * 32 + hb * 16);
#pragma unroll
    for (int mf = 0; mf < 4; ++mf)
#pragma unroll
      for (int jf = 0; jf < 4; ++jf)
        acc[mf][jf] = __builtin_amdgcn_wmma_f32_16x16x32_bf16(
            false, af[mf], false, bfm[jf], (short)0, acc[mf][jf], false, false);
  };

  zero_acc();

#if HAVE_TDM
  const bool issuer = (wave == 0);
  // A tile: 64 rows x 32 bf16 (8 x 8B units), pitch 512 bf16 = 128 units
  auto issueA = [&](int kt, int buf) {
    tdm_load_2d(qb + (size_t)m_base * D + kt * 32,
                (uint32_t)(uintptr_t)&As[buf][0], 64u, 8u, 128u);
  };
  auto issueB = [&](const bf16_t* W, int kt, int buf) {
    tdm_load_2d(W + kt * 32, (uint32_t)(uintptr_t)&Bs[buf][0], 512u, 8u, 128u);
  };

  // ---- GEMM1: H = q * W1  (TDM double-buffered, 1 barrier / k-step) ------
  if (issuer) { issueA(0, 0); issueB(W1s, 0, 0); }
  for (int kt = 0; kt < 16; ++kt) {
    int cur = kt & 1;
    if (issuer) __builtin_amdgcn_s_wait_tensorcnt(0);
    __syncthreads();                       // tile[cur] ready, tile[1-cur] free
    if (issuer && kt + 1 < 16) { issueA(kt + 1, 1 ^ cur); issueB(W1s, kt + 1, 1 ^ cur); }
    mma_step(&As[cur][0], &Bs[cur][0]);    // overlaps next tile's DMA
  }
#else
  // fallback: synchronous staging through VGPRs
  for (int kt = 0; kt < 16; ++kt) {
    {
      int row = tid >> 2, ch = (tid & 3) * 8;
      *(v8bf*)(&As[0][0] + row * 32 + ch) =
          *(const v8bf*)(qb + (size_t)(m_base + row) * D + kt * 32 + ch);
    }
#pragma unroll
    for (int i = 0; i < 8; ++i) {
      int c = tid + i * 256;
      int N = c >> 2, ch = (c & 3) * 8;
      *(v8bf*)(&Bs[0][0] + N * 32 + ch) =
          *(const v8bf*)(W1s + (size_t)N * D + kt * 32 + ch);
    }
    __syncthreads();
    mma_step(&As[0][0], &Bs[0][0]);
    __syncthreads();
  }
#endif

  // ---- bias + LeakyReLU -> bf16 H tile in LDS ---------------------------
#pragma unroll
  for (int jf = 0; jf < 4; ++jf) {
    int j = n0 + jf * 16 + l15;
    float bia = b1[slot * D + j];
#pragma unroll
    for (int mf = 0; mf < 4; ++mf)
#pragma unroll
      for (int r = 0; r < 8; ++r) {
        float v = acc[mf][jf][r] + bia;
        v = (v >= 0.f) ? v : v * slope;
        Hs[(mf * 16 + hb * 8 + r) * D + j] = (__bf16)v;
      }
  }
  if (tid < 64) Lg[tid] = 0.f;
  __syncthreads();   // H complete; all GEMM1 B reads done -> Bs reusable

  // ---- GEMM2: X = tanh(H * W2 + b2), feed gate --------------------------
  zero_acc();
#if HAVE_TDM
  if (issuer) issueB(W2s, 0, 0);
  for (int kt = 0; kt < 16; ++kt) {
    int cur = kt & 1;
    if (issuer) __builtin_amdgcn_s_wait_tensorcnt(0);
    __syncthreads();
    if (issuer && kt + 1 < 16) issueB(W2s, kt + 1, 1 ^ cur);
    mma_step_h(kt, &Bs[cur][0]);
  }
#else
  for (int kt = 0; kt < 16; ++kt) {
#pragma unroll
    for (int i = 0; i < 8; ++i) {
      int c = tid + i * 256;
      int N = c >> 2, ch = (c & 3) * 8;
      *(v8bf*)(&Bs[0][0] + N * 32 + ch) =
          *(const v8bf*)(W2s + (size_t)N * D + kt * 32 + ch);
    }
    __syncthreads();
    mma_step_h(kt, &Bs[0][0]);
    __syncthreads();
  }
#endif

  // gate: logit[m] += sum_j tanh(x)[m,j] * Wg[j]
  float part[4][8];
#pragma unroll
  for (int mf = 0; mf < 4; ++mf)
#pragma unroll
    for (int r = 0; r < 8; ++r) part[mf][r] = 0.f;
#pragma unroll
  for (int jf = 0; jf < 4; ++jf) {
    int j = n0 + jf * 16 + l15;
    float b2v = b2[slot * D + j];
    float wgv = Wg[slot * D + j];
#pragma unroll
    for (int mf = 0; mf < 4; ++mf)
#pragma unroll
      for (int r = 0; r < 8; ++r)
        part[mf][r] += tanhf(acc[mf][jf][r] + b2v) * wgv;
  }
#pragma unroll
  for (int mf = 0; mf < 4; ++mf)
#pragma unroll
    for (int r = 0; r < 8; ++r) {
      float v = part[mf][r];
      v += __shfl_xor(v, 1, 32);
      v += __shfl_xor(v, 2, 32);
      v += __shfl_xor(v, 4, 32);
      v += __shfl_xor(v, 8, 32);          // reduce across the 16 columns
      if (l15 == 0) atomicAdd(&Lg[mf * 16 + hb * 8 + r], v);  // ds_add_f32
    }
  __syncthreads();
  if (tid < 64) logits[(size_t)slot * ROWS + m_base + tid] = Lg[tid];
}

// ---------------- Pass B: softmax stats + top-8 per (slot, b) --------------
__global__ __launch_bounds__(256)
void k_topk(const float* __restrict__ logits,
            float* __restrict__ topw, int* __restrict__ topi) {
  __shared__ float sv[256];
  __shared__ int   si[256];
  int slot = blockIdx.x >> 2, b = blockIdx.x & 3;
  const float* lg = logits + (size_t)slot * ROWS + b * LSEQ;
  int tid = threadIdx.x;

  float lv[16];
#pragma unroll
  for (int i = 0; i < 16; ++i) lv[i] = lg[tid + i * 256];

  float m = -3.4e38f;
#pragma unroll
  for (int i = 0; i < 16; ++i) m = fmaxf(m, lv[i]);
  sv[tid] = m; __syncthreads();
  for (int s = 128; s > 0; s >>= 1) {
    if (tid < s) sv[tid] = fmaxf(sv[tid], sv[tid + s]);
    __syncthreads();
  }
  float rmax = sv[0]; __syncthreads();

  float ssum = 0.f;
#pragma unroll
  for (int i = 0; i < 16; ++i) ssum += expf(lv[i] - rmax);
  sv[tid] = ssum; __syncthreads();
  for (int s = 128; s > 0; s >>= 1) {
    if (tid < s) sv[tid] += sv[tid + s];
    __syncthreads();
  }
  float denom = sv[0]; __syncthreads();

  for (int k = 0; k < TOPK; ++k) {
    float bv = -3.4e38f; int bi = 0x7fffffff;
#pragma unroll
    for (int i = 0; i < 16; ++i)
      if (lv[i] > bv) { bv = lv[i]; bi = tid + i * 256; }
    sv[tid] = bv; si[tid] = bi; __syncthreads();
    for (int s = 128; s > 0; s >>= 1) {
      if (tid < s) {
        float ov = sv[tid + s]; int oi = si[tid + s];
        if (ov > sv[tid] || (ov == sv[tid] && oi < si[tid])) {
          sv[tid] = ov; si[tid] = oi;
        }
      }
      __syncthreads();
    }
    float wv = sv[0]; int wi = si[0]; __syncthreads();
    if (tid == 0) {
      topw[(slot * NB + b) * TOPK + k] = expf(wv - rmax) / denom;
      topi[(slot * NB + b) * TOPK + k] = wi;
    }
    if ((wi & 255) == tid) lv[wi >> 8] = -3.4e38f;  // mask the winner
    __syncthreads();
  }
}

// ---------------- Pass C: recompute top-k x rows, combine, normalize -------
__global__ __launch_bounds__(256)
void k_combine(const float* __restrict__ query,
               const bf16_t* __restrict__ w1t, const bf16_t* __restrict__ w2t,
               const float* __restrict__ b1,  const float* __restrict__ b2,
               const float* __restrict__ topw, const int* __restrict__ topi,
               float* __restrict__ out) {
  __shared__ float qs[D], hsm[D], comb[D], red[256];
  int slot = blockIdx.x, b = blockIdx.y, tid = threadIdx.x;
  float slope = 0.01f + 0.0125f * (float)slot;
  comb[tid] = 0.f; comb[tid + 256] = 0.f;

  for (int k = 0; k < TOPK; ++k) {
    int   idx = topi[(slot * NB + b) * TOPK + k];
    float w   = topw[(slot * NB + b) * TOPK + k];
    const float* qr = query + ((size_t)b * LSEQ + idx) * D;
    qs[tid] = qr[tid]; qs[tid + 256] = qr[tid + 256];
    __syncthreads();

    float hv[2];
#pragma unroll
    for (int half = 0; half < 2; ++half) {
      int j = tid + half * 256;
      const bf16_t* wr = w1t + ((size_t)slot * D + j) * D;
      float a = b1[slot * D + j];
      for (int d0 = 0; d0 < D; d0 += 8) {
        v8bf wv = *(const v8bf*)(wr + d0);
#pragma unroll
        for (int i = 0; i < 8; ++i) a += qs[d0 + i] * (float)wv[i];
      }
      hv[half] = (a >= 0.f) ? a : a * slope;
    }
    hsm[tid] = hv[0]; hsm[tid + 256] = hv[1];
    __syncthreads();

#pragma unroll
    for (int half = 0; half < 2; ++half) {
      int j = tid + half * 256;
      const bf16_t* wr = w2t + ((size_t)slot * D + j) * D;
      float a = b2[slot * D + j];
      for (int d0 = 0; d0 < D; d0 += 8) {
        v8bf wv = *(const v8bf*)(wr + d0);
#pragma unroll
        for (int i = 0; i < 8; ++i) a += hsm[d0 + i] * (float)wv[i];
      }
      comb[j] += w * tanhf(a);
    }
    __syncthreads();
  }

  float s = comb[tid] * comb[tid] + comb[tid + 256] * comb[tid + 256];
  red[tid] = s; __syncthreads();
  for (int st = 128; st > 0; st >>= 1) {
    if (tid < st) red[tid] += red[tid + st];
    __syncthreads();
  }
  float rn = 1.f / fmaxf(sqrtf(red[0]), 1e-12f);
  out[((size_t)b * NMEM + slot) * D + tid]       = comb[tid] * rn;
  out[((size_t)b * NMEM + slot) * D + tid + 256] = comb[tid + 256] * rn;
}

// ---------------------------------------------------------------------------
extern "C" void kernel_launch(void* const* d_in, const int* in_sizes, int n_in,
                              void* d_out, int out_size, void* d_ws, size_t ws_size,
                              hipStream_t stream) {
  (void)in_sizes; (void)n_in; (void)out_size; (void)ws_size;
  const float* query = (const float*)d_in[0];
  const float* W1    = (const float*)d_in[1];
  const float* b1    = (const float*)d_in[2];
  const float* W2    = (const float*)d_in[3];
  const float* b2    = (const float*)d_in[4];
  const float* Wg    = (const float*)d_in[5];
  // d_in[6] = bg: constant per slot over L -> softmax-invariant, unused.
  // d_in[7] = topk scalar (== 8, compile-time TOPK).

  char* ws = (char*)d_ws;
  bf16_t* qb     = (bf16_t*)(ws);                               // 16 MB
  bf16_t* w1t    = (bf16_t*)(ws + 16777216);                    //  8 MB
  bf16_t* w2t    = (bf16_t*)(ws + 16777216 + 8388608);          //  8 MB
  float*  logits = (float*) (ws + 33554432);                    //  1 MB
  float*  topw   = (float*) (ws + 34603008);
  int*    topi   = (int*)   (ws + 34605056);
  float*  out    = (float*)d_out;

  k_cvt_query  <<<8192,           256, 0, stream>>>(query, qb);
  k_transpose_w<<<dim3(16,16,32), 256, 0, stream>>>(W1, W2, w1t, w2t);
  k_fused_logits<<<dim3(256,16),  256, 0, stream>>>(qb, w1t, w2t, b1, b2, Wg, logits);
  k_topk       <<<64,             256, 0, stream>>>(logits, topw, topi);
  k_combine    <<<dim3(16,4),     256, 0, stream>>>(query, w1t, w2t, b1, b2, topw, topi, out);
}
